// QuantizationLayer_39702677684765
// MI455X (gfx1250) — matrix operand order repaired and checked by reading, not compiled, not run on hardware
//
#include <hip/hip_runtime.h>

// Problem constants (match reference)
#define BB 65536
#define LL 4
#define KK 256
#define DD 128

#define ROWS 64           // rows per block (4 waves x 16 rows)
#define THREADS 128
#define NBLK (BB / ROWS)  // 1024 blocks
#define PITCH (DD + 4)    // LDS pitch: 132 floats -> conflict-free row access

typedef __attribute__((ext_vector_type(16))) __bf16 bf16x16;
typedef __attribute__((ext_vector_type(8)))  float  v8f;

__device__ __forceinline__ v8f wmma_bf16(bf16x16 a, bf16x16 b, v8f c) {
  // D = A(16x32 bf16) * B(32x16 bf16) + C(16x16 f32)
  return __builtin_amdgcn_wmma_f32_16x16x32_bf16(false, a, false, b, (short)0, c,
                                                 false, false);
}

// ---------------------------------------------------------------------------
// Prep: split codebooks into bf16 hi/lo, compute ||cb||^2, zero counts.
// grid = L*K blocks, 128 threads (one dim element each)
// ---------------------------------------------------------------------------
__global__ __launch_bounds__(128) void rvq_prep(const float* __restrict__ cb,
                                                __bf16* __restrict__ cb_hi,
                                                __bf16* __restrict__ cb_lo,
                                                float* __restrict__ cbsq,
                                                int* __restrict__ counts) {
  __shared__ float s[128];
  const int c = blockIdx.x;          // 0 .. L*K-1
  const int d = threadIdx.x;         // 0 .. 127
  const size_t o = (size_t)c * DD + d;
  float w = cb[o];
  __bf16 h = (__bf16)w;
  cb_hi[o] = h;
  cb_lo[o] = (__bf16)(w - (float)h);
  s[d] = w * w;
  __syncthreads();
  for (int off = 64; off > 0; off >>= 1) {
    if (d < off) s[d] += s[d + off];
    __syncthreads();
  }
  if (d == 0) {
    cbsq[c] = s[0];
    counts[c] = 0;
  }
}

// ---------------------------------------------------------------------------
// Fused residual-VQ over all 4 levels. One block owns 64 rows kept in LDS.
// ---------------------------------------------------------------------------
__global__ __launch_bounds__(THREADS) void rvq_fused(
    const float* __restrict__ x, const float* __restrict__ cb32,
    const __bf16* __restrict__ cb_hi, const __bf16* __restrict__ cb_lo,
    const float* __restrict__ cbsq, float* __restrict__ col_part,
    int* __restrict__ counts, float* __restrict__ out_idx,
    float* __restrict__ out_r, float* __restrict__ out_e,
    float* __restrict__ out_z) {
  __shared__ float s_res[ROWS * PITCH];
  __shared__ float s_colsum[KK];
  __shared__ float s_cbsq[KK];
  __shared__ int s_codes[ROWS];

  const int tid = threadIdx.x;
  const int lane = tid & 31;
  const int wave = tid >> 5;   // 0..3
  const int m = lane & 15;     // column-in-tile / row-in-tile id
  const int half = lane >> 4;  // 0 or 1
  const int rowBase = blockIdx.x * ROWS;
  const int wr = wave * 16;    // wave's row base inside block

  // Stage x tile into LDS (coalesced).
  for (int i = tid; i < ROWS * DD; i += THREADS) {
    int r = i >> 7, d = i & (DD - 1);
    s_res[r * PITCH + d] = x[(size_t)(rowBase + r) * DD + d];
  }
  __syncthreads();

  for (int l = 0; l < LL; ++l) {
    // ---- A-matrix conversion: fp32 residual -> bf16 hi/lo in WMMA A layout.
    // k order used here, not AMD's table: lanes 0-15 row m hold K {0..15}; lanes 16-31 hold
    // K {16..31} of the same rows.
    bf16x16 ah[4], al[4];
    float sq = 0.f;
#pragma unroll
    for (int ks = 0; ks < 4; ++ks) {
      const float* src = &s_res[(wr + m) * PITCH + ks * 32 + half * 16];
#pragma unroll
      for (int e = 0; e < 8; ++e) {
        float f0 = src[e];
        float f1 = src[8 + e];
        sq += f0 * f0 + f1 * f1;
        __bf16 h0 = (__bf16)f0;
        ah[ks][e] = h0;
        al[ks][e] = (__bf16)(f0 - (float)h0);
        __bf16 h1 = (__bf16)f1;
        ah[ks][8 + e] = h1;
        al[ks][8 + e] = (__bf16)(f1 - (float)h1);
      }
    }
    // ||res||^2 per row: lane m and lane m+16 each hold half the row.
    float rowSq = sq + __shfl_xor(sq, 16, 32);
    // Distribute to the C-layout rows this lane will produce (half*8+v).
    float rs[8];
#pragma unroll
    for (int v = 0; v < 8; ++v) rs[v] = __shfl(rowSq, half * 8 + v, 32);

    for (int i = tid; i < KK; i += THREADS) {
      s_colsum[i] = 0.f;
      s_cbsq[i] = cbsq[l * KK + i];
    }
    __syncthreads();

    float bd[8];
    int bc[8];
#pragma unroll
    for (int v = 0; v < 8; ++v) {
      bd[v] = 3.4e38f;
      bc[v] = 0;
    }

    // ---- 16 column tiles of 16 codes; D=128 in 4 k-steps of 32.
    for (int ct = 0; ct < 16; ++ct) {
      v8f acc = {0.f, 0.f, 0.f, 0.f, 0.f, 0.f, 0.f, 0.f};
      const int code = ct * 16 + m;
      // B in A's k order (not AMD's table): lane = column (code), 16 consecutive K per lane,
      // halves split K 0..15 / 16..31 -> contiguous 32B per lane.
      const size_t bbase = ((size_t)l * KK + code) * DD + half * 16;
#pragma unroll
      for (int ks = 0; ks < 4; ++ks) {
        bf16x16 bh = *(const bf16x16*)(cb_hi + bbase + ks * 32);
        bf16x16 bl = *(const bf16x16*)(cb_lo + bbase + ks * 32);
        acc = wmma_bf16(ah[ks], bh, acc);  // hi*hi
        acc = wmma_bf16(ah[ks], bl, acc);  // hi*lo
        acc = wmma_bf16(al[ks], bh, acc);  // lo*hi
      }
      // Epilogue: d2 = ||r||^2 - 2 r.cb + ||cb||^2 ; dist = sqrt(max(d2,0))
      const float cbs = s_cbsq[code];
      float csum = 0.f;
#pragma unroll
      for (int v = 0; v < 8; ++v) {
        float d2 = rs[v] - 2.0f * acc[v] + cbs;
        float dist = __builtin_sqrtf(fmaxf(d2, 0.f));
        csum += dist;
        if (dist < bd[v]) {  // codes ascend across ct -> '<' keeps first index
          bd[v] = dist;
          bc[v] = code;
        }
      }
      atomicAdd(&s_colsum[code], csum);
    }

    // ---- cross-lane argmin over the 16 lanes of each half (xor masks 1..8).
#pragma unroll
    for (int off = 8; off >= 1; off >>= 1) {
#pragma unroll
      for (int v = 0; v < 8; ++v) {
        float od = __shfl_xor(bd[v], off, 32);
        int oc = __shfl_xor(bc[v], off, 32);
        if (od < bd[v] || (od == bd[v] && oc < bc[v])) {
          bd[v] = od;
          bc[v] = oc;
        }
      }
    }
    if (m == 0) {  // lane 0 -> rows 0..7, lane 16 -> rows 8..15
#pragma unroll
      for (int v = 0; v < 8; ++v) {
        int r = wr + half * 8 + v;
        int grow = rowBase + r;
        s_codes[r] = bc[v];
        out_idx[(size_t)grow * LL + l] = (float)bc[v];
        atomicAdd(&counts[l * KK + bc[v]], 1);
      }
    }
    __syncthreads();

    // Deterministic column-sum partials (no float atomics to global).
    for (int i = tid; i < KK; i += THREADS)
      col_part[((size_t)l * NBLK + blockIdx.x) * KK + i] = s_colsum[i];

    // Outputs + residual update (coalesced; codebook gather hits L2).
    for (int i = tid; i < ROWS * DD; i += THREADS) {
      int r = i >> 7, d = i & (DD - 1);
      float rv = s_res[r * PITCH + d];
      int code = s_codes[r];
      float q = cb32[((size_t)l * KK + code) * DD + d];
      size_t o = ((size_t)(rowBase + r) * LL + l) * DD + d;
      out_r[o] = rv;
      out_e[o] = q;
      s_res[r * PITCH + d] = rv - q;
    }
    __syncthreads();
  }

  // z_hat = x - res_final  (== sum of quantized vectors)
  for (int i = tid; i < ROWS * DD; i += THREADS) {
    int r = i >> 7, d = i & (DD - 1);
    size_t g = (size_t)(rowBase + r) * DD + d;
    out_z[g] = x[g] - s_res[r * PITCH + d];
  }
}

// ---------------------------------------------------------------------------
// Finalize: flops_loss = sum_l sum_k (mean_b dist)^2 ; count = #unused codes.
// ---------------------------------------------------------------------------
__global__ __launch_bounds__(256) void rvq_final(
    const float* __restrict__ col_part, const int* __restrict__ counts,
    float* __restrict__ out_count, float* __restrict__ out_flops) {
  __shared__ float sf[256];
  __shared__ int si[256];
  const int t = threadIdx.x;
  float fa = 0.f;
  int ca = 0;
  for (int i = t; i < LL * KK; i += 256) {
    int l = i >> 8, k = i & (KK - 1);
    float s = 0.f;
    for (int b = 0; b < NBLK; ++b)
      s += col_part[((size_t)l * NBLK + b) * KK + k];
    float mean = s * (1.0f / (float)BB);
    fa += mean * mean;
    ca += (counts[i] < 1) ? 1 : 0;
  }
  sf[t] = fa;
  si[t] = ca;
  __syncthreads();
  for (int off = 128; off > 0; off >>= 1) {
    if (t < off) {
      sf[t] += sf[t + off];
      si[t] += si[t + off];
    }
    __syncthreads();
  }
  if (t == 0) {
    *out_count = (float)si[0];
    *out_flops = sf[0];
  }
}

// ---------------------------------------------------------------------------
extern "C" void kernel_launch(void* const* d_in, const int* in_sizes, int n_in,
                              void* d_out, int out_size, void* d_ws,
                              size_t ws_size, hipStream_t stream) {
  const float* x = (const float*)d_in[0];         // (B, D) fp32
  const float* cb = (const float*)d_in[1];        // (L, K, D) fp32

  // Workspace carve (256B aligned)
  char* w = (char*)d_ws;
  auto carve = [&](size_t bytes) {
    void* p = (void*)w;
    w += (bytes + 255) & ~(size_t)255;
    return p;
  };
  __bf16* cb_hi = (__bf16*)carve((size_t)LL * KK * DD * sizeof(__bf16));
  __bf16* cb_lo = (__bf16*)carve((size_t)LL * KK * DD * sizeof(__bf16));
  float* cbsq = (float*)carve((size_t)LL * KK * sizeof(float));
  int* counts = (int*)carve((size_t)LL * KK * sizeof(int));
  float* col_part = (float*)carve((size_t)LL * NBLK * KK * sizeof(float));
  (void)ws_size;
  (void)n_in;
  (void)in_sizes;
  (void)out_size;

  // Output carve (tuple concatenated flat, float)
  float* out = (float*)d_out;
  float* out_idx = out;                                // (B, L)
  float* out_r = out_idx + (size_t)BB * LL;            // (B, L, D)
  float* out_e = out_r + (size_t)BB * LL * DD;         // (B, L, D)
  float* out_z = out_e + (size_t)BB * LL * DD;         // (B, D)
  float* out_cnt = out_z + (size_t)BB * DD;            // scalar
  float* out_fl = out_cnt + 1;                         // scalar

  rvq_prep<<<LL * KK, 128, 0, stream>>>(cb, cb_hi, cb_lo, cbsq, counts);
  rvq_fused<<<NBLK, THREADS, 0, stream>>>(x, cb, cb_hi, cb_lo, cbsq, col_part,
                                          counts, out_idx, out_r, out_e, out_z);
  rvq_final<<<1, 256, 0, stream>>>(col_part, counts, out_cnt, out_fl);
}
